// LinearRNN_41506563948811
// MI455X (gfx1250) — matrix operand order repaired and checked by reading, not compiled, not run on hardware
//
#include <hip/hip_runtime.h>
#include <stdint.h>

// ---------------------------------------------------------------------------
// Types for CDNA5 WMMA (wave32)
// ---------------------------------------------------------------------------
typedef __bf16 bf16_t;
typedef bf16_t v16bf __attribute__((ext_vector_type(16)));
typedef float  v8f   __attribute__((ext_vector_type(8)));
typedef unsigned int u32x4 __attribute__((ext_vector_type(4)));
typedef unsigned int u32x8 __attribute__((ext_vector_type(8)));
typedef int i32x4 __attribute__((ext_vector_type(4)));
typedef int i32x8 __attribute__((ext_vector_type(8)));

// Problem sizes (fixed by the reference)
#define BATCH 32
#define TLEN  1024
#define INDIM 256
#define HDIM  512
#define NCHUNK 64
#define LCHUNK 16   // TLEN / NCHUNK; combine uses Whh^16

// Workspace layout, in dwords:
//   whh_img @ 0        : 131072   (512x512 bf16 B-fragment image)
//   wxh_img @ 131072   : 65536    (512x256 bf16 B-fragment image)
//   xbf     @ 196608   : 4194304  (x converted to bf16, same [B,T,IN] layout)
//   w16_img @ 4390912  : 131072   (Whh^16 bf16 B-fragment image)
//   p0      @ 4521984  : 262144   (f32, W^2 / W^8)
//   p1      @ 4784128  : 262144   (f32, W^4 / W^16)
//   v       @ 5046272  : 1048576  (f32, chunk results)
#define WS_WXH_IMG 131072
#define WS_XBF     196608
#define WS_W16_IMG 4390912
#define WS_P0      4521984
#define WS_P1      4784128
#define WS_V       5046272
#define WS_FULL_DW 6094848
#define WS_MID_DW  4390912   // whh_img + wxh_img + xbf

// Opaque zero offset: defeats LICM (so streamed fragment loads stay inside
// the sequential loop instead of being hoisted + spilled to scratch) while
// keeping the base pointer's global address-space provenance intact, so the
// loads still lower to global_load_b128 (flat would also tie up DScnt).
__device__ __forceinline__ unsigned opaque_zero() {
  unsigned z = 0;
  asm volatile("" : "+v"(z));
  return z;
}

// ---------------------------------------------------------------------------
// Tensor Data Mover: async 2D tile (tile_h x tile_w elements) global -> LDS.
// dsize_code: 0=1B, 1=2B, 2=4B, 3=8B.  Descriptor per CDNA5 ISA.
// ---------------------------------------------------------------------------
__device__ __forceinline__ void tdm_load_2d(unsigned lds_byte_off,
                                            const void* gsrc,
                                            unsigned tile_w, unsigned tile_h,
                                            unsigned row_stride_elts,
                                            unsigned dsize_code) {
#if __has_builtin(__builtin_amdgcn_tensor_load_to_lds)
  unsigned long long ga = (unsigned long long)(uintptr_t)gsrc;
  u32x4 g0; i32x8 g1; i32x4 g2; i32x4 g3;
  g0[0] = 1u;                                   // count=1
  g0[1] = lds_byte_off;                         // lds_addr
  g0[2] = (unsigned)(ga & 0xffffffffull);       // global_addr lo
  g0[3] = (unsigned)((ga >> 32) & 0x01ffffffull) | (2u << 30);  // hi | type=2
  g1[0] = (int)(dsize_code << 16);
  g1[1] = (int)((tile_w & 0xffffu) << 16);
  g1[2] = (int)(((tile_w >> 16) & 0xffffu) | ((tile_h & 0xffffu) << 16));
  g1[3] = (int)(((tile_h >> 16) & 0xffffu) | ((tile_w & 0xffffu) << 16));
  g1[4] = (int)(tile_h & 0xffffu);
  g1[5] = (int)row_stride_elts;
  g1[6] = 0; g1[7] = 0;
  g2[0] = g2[1] = g2[2] = g2[3] = 0;
  g3[0] = g3[1] = g3[2] = g3[3] = 0;
#if defined(__clang_major__) && (__clang_major__ >= 23)
  i32x8 g4; g4[0]=g4[1]=g4[2]=g4[3]=g4[4]=g4[5]=g4[6]=g4[7]=0;
  __builtin_amdgcn_tensor_load_to_lds(g0, g1, g2, g3, g4, 0);
#else
  __builtin_amdgcn_tensor_load_to_lds(g0, g1, g2, g3, 0);
#endif
#else
  (void)lds_byte_off; (void)gsrc; (void)tile_w; (void)tile_h;
  (void)row_stride_elts; (void)dsize_code;
#endif
}

__device__ __forceinline__ void tdm_wait0() {
#if __has_builtin(__builtin_amdgcn_s_wait_tensorcnt)
  __builtin_amdgcn_s_wait_tensorcnt(0);
#endif
}

// ---------------------------------------------------------------------------
// WMMA helpers
// A (16x32, MxK): lane m = lane&15, kb = (lane>>4)*8; elems {kb..kb+7, kb+16..kb+23}
// B (32x16, KxN): lane n = lane&15, kb = (lane>>4)*8; elems {kb..kb+7, kb+16..kb+23}
// ---------------------------------------------------------------------------
// A-fragment straight from bf16 memory (LDS): two 16B loads, no conversion.
__device__ __forceinline__ v16bf frag_a_bf(const bf16_t* p) {
  const u32x4 a = *(const u32x4*)p;         // elems 0..7  (k = kb..kb+7)
  const u32x4 b = *(const u32x4*)(p + 16);  // elems 8..15 (k = kb+16..kb+23)
  u32x8 r;
  r[0]=a[0]; r[1]=a[1]; r[2]=a[2]; r[3]=a[3];
  r[4]=b[0]; r[5]=b[1]; r[6]=b[2]; r[7]=b[3];
  return __builtin_bit_cast(v16bf, r);
}

// B-fragment from a pre-packed fragment image (32B per lane).
__device__ __forceinline__ v16bf frag_b_img(const unsigned* p) {
  u32x8 raw = *(const u32x8*)p;
  return __builtin_bit_cast(v16bf, raw);
}

// A-fragment from f32 memory with conversion (used only in matsq).
__device__ __forceinline__ v16bf frag_a_f32(const float* p) {
  const float4 q0 = *(const float4*)(p + 0);
  const float4 q1 = *(const float4*)(p + 4);
  const float4 q2 = *(const float4*)(p + 16);
  const float4 q3 = *(const float4*)(p + 20);
  v16bf f;
  f[0]=(bf16_t)q0.x; f[1]=(bf16_t)q0.y; f[2]=(bf16_t)q0.z; f[3]=(bf16_t)q0.w;
  f[4]=(bf16_t)q1.x; f[5]=(bf16_t)q1.y; f[6]=(bf16_t)q1.z; f[7]=(bf16_t)q1.w;
  f[8]=(bf16_t)q2.x; f[9]=(bf16_t)q2.y; f[10]=(bf16_t)q2.z; f[11]=(bf16_t)q2.w;
  f[12]=(bf16_t)q3.x; f[13]=(bf16_t)q3.y; f[14]=(bf16_t)q3.z; f[15]=(bf16_t)q3.w;
  return f;
}

__device__ __forceinline__ v8f wmma_bf16(v16bf a, v16bf b, v8f c) {
  return __builtin_amdgcn_wmma_f32_16x16x32_bf16(false, a, false, b, (short)0, c,
                                                 false, false);
}

__device__ __forceinline__ unsigned pack2bf(float lo, float hi) {
  unsigned a = (unsigned)__builtin_bit_cast(unsigned short, (bf16_t)lo);
  unsigned b = (unsigned)__builtin_bit_cast(unsigned short, (bf16_t)hi);
  return a | (b << 16);
}

// ---------------------------------------------------------------------------
// Pack W (row-major [512 x K] f32) into a bf16 WMMA B-fragment image:
//   img[((ntile*(K/32) + kc)*32 + lane)*8 + d]
// ---------------------------------------------------------------------------
__global__ __launch_bounds__(256) void
pack_b_image(const float* __restrict__ W, unsigned* __restrict__ img, int K) {
  const int id = blockIdx.x * 256 + threadIdx.x;
  const int nkc = K / 32;
  if (id >= 32 * nkc * 32) return;
  const int lane = id & 31;
  const int kc   = (id >> 5) % nkc;
  const int tile = (id >> 5) / nkc;
  const int n  = tile * 16 + (lane & 15);
  const int k0 = kc * 32 + (lane >> 4) * 8;
  const float* p = W + (size_t)n * K + k0;
  u32x4 d0, d1;
#pragma unroll
  for (int j = 0; j < 4; ++j) d0[j] = pack2bf(p[2*j],     p[2*j+1]);
#pragma unroll
  for (int j = 0; j < 4; ++j) d1[j] = pack2bf(p[16 + 2*j], p[16 + 2*j+1]);
  unsigned* dst = img + (size_t)id * 8;
  *(u32x4*)dst       = d0;
  *(u32x4*)(dst + 4) = d1;
}

// Convert x [B*T*IN f32] -> bf16 (packed pairs), same element order.
__global__ __launch_bounds__(256) void
pack_x_bf16(const float* __restrict__ x, unsigned* __restrict__ xbf, int ndw) {
  const int id = blockIdx.x * 256 + threadIdx.x;
  if (id >= ndw) return;
  const float2 q = *(const float2*)(x + (size_t)id * 2);
  xbf[id] = pack2bf(q.x, q.y);
}

// ---------------------------------------------------------------------------
// Phase A: per-chunk fused scan.  One workgroup (32 waves) per chunk; wave w
// owns output N-tile w.  All operands are bf16 at rest: weight B-fragments
// stream from packed images in L2 (32B/lane global_load_b128s, re-fetched
// every step via an opaque offset so they are neither LICM-hoisted into
// scratch spills nor demoted to flat loads), h state is bf16 in LDS, x_t
// bf16 tiles are TDM double-buffered.  Final step writes f32 accumulators
// straight to vout.
// ---------------------------------------------------------------------------
__global__ __launch_bounds__(1024, 1) void
chunk_scan_kernel(const unsigned* __restrict__ xbf,     // [B, T, IN] bf16 pairs
                  const unsigned* __restrict__ wxh_img, // packed Wxh fragments
                  const float* __restrict__ bxh,        // [H]
                  const unsigned* __restrict__ whh_img, // packed Whh fragments
                  float* __restrict__ vout,             // [gridDim.x, B, H]
                  int Lsteps) {
  __shared__ bf16_t xbuf[2][BATCH * INDIM];  // 2 x 16 KB
  __shared__ bf16_t hbuf[BATCH * HDIM];      // 32 KB
  const int tid  = threadIdx.x;
  const int wave = tid >> 5;        // N-tile index, 0..31
  const int lane = tid & 31;
  const int lo   = lane & 15;
  const int hi   = lane >> 4;
  const int ncol = wave * 16 + lo;
  const int chunk = blockIdx.x;

  const unsigned* wbase = whh_img + ((size_t)(wave * (HDIM / 32)) * 32 + lane) * 8;
  const unsigned* xbase = wxh_img + ((size_t)(wave * (INDIM / 32)) * 32 + lane) * 8;
  const float bias = bxh[ncol];

  {  // h_0 = 0
    unsigned* hz = (unsigned*)&hbuf[0];
    for (int i = tid; i < BATCH * HDIM / 2; i += 1024) hz[i] = 0u;
  }

  if (wave == 0) {  // prime x_0 via TDM (bf16 tile: data_size code 1)
    tdm_load_2d((unsigned)(uintptr_t)(void*)&xbuf[0][0],
                xbf + ((size_t)chunk * Lsteps * INDIM) / 2, INDIM, BATCH,
                (unsigned)TLEN * INDIM, 1u);
    tdm_wait0();
  }
  __syncthreads();

  for (int t = 0; t < Lsteps; ++t) {
    if (wave == 0 && (t + 1) < Lsteps) {  // prefetch x_{t+1}
      tdm_load_2d((unsigned)(uintptr_t)(void*)&xbuf[(t + 1) & 1][0],
                  xbf + ((size_t)(chunk * Lsteps + t + 1) * INDIM) / 2, INDIM,
                  BATCH, (unsigned)TLEN * INDIM, 1u);
    }

    // Loop-variant (to the compiler) zero offset: keeps the 24 B-fragment
    // global loads below inside the loop, as global_load_b128 from L2.
    const unsigned zoff = opaque_zero();
    const unsigned* wb = wbase + zoff;
    const unsigned* xw = xbase + zoff;

    v8f acc0, acc1;  // M-tiles 0 (rows 0..15) and 1 (rows 16..31)
#pragma unroll
    for (int r = 0; r < 8; ++r) { acc0[r] = bias; acc1[r] = bias; }

    // u_t: x_t [32x256] @ Wxh^T
    const bf16_t* xb = &xbuf[t & 1][0];
#pragma unroll
    for (int kc = 0; kc < INDIM / 32; ++kc) {
      v16bf bx = frag_b_img(xw + (size_t)kc * 256);
      v16bf a0 = frag_a_bf(xb + (size_t)lo * INDIM + kc * 32 + hi * 8);
      v16bf a1 = frag_a_bf(xb + (size_t)(lo + 16) * INDIM + kc * 32 + hi * 8);
      acc0 = wmma_bf16(a0, bx, acc0);
      acc1 = wmma_bf16(a1, bx, acc1);
    }
    // recurrence: + h_{t-1} [32x512] @ Whh^T
#pragma unroll
    for (int kc = 0; kc < HDIM / 32; ++kc) {
      v16bf bw = frag_b_img(wb + (size_t)kc * 256);
      v16bf a0 = frag_a_bf(hbuf + (size_t)lo * HDIM + kc * 32 + hi * 8);
      v16bf a1 = frag_a_bf(hbuf + (size_t)(lo + 16) * HDIM + kc * 32 + hi * 8);
      acc0 = wmma_bf16(a0, bw, acc0);
      acc1 = wmma_bf16(a1, bw, acc1);
    }

    if (wave == 0 && (t + 1) < Lsteps) tdm_wait0();
    __syncthreads();  // everyone finished reading h_{t-1}
    if (t == Lsteps - 1) {
      // final step: full-precision f32 result straight to global
      float* vc = vout + (size_t)chunk * (BATCH * HDIM);
#pragma unroll
      for (int r = 0; r < 8; ++r) {  // C/D layout: M = r + 8*hi, N = ncol
        vc[(size_t)(r + hi * 8) * HDIM + ncol]      = acc0[r];
        vc[(size_t)(16 + r + hi * 8) * HDIM + ncol] = acc1[r];
      }
    } else {
#pragma unroll
      for (int r = 0; r < 8; ++r) {
        hbuf[(size_t)(r + hi * 8) * HDIM + ncol]      = (bf16_t)acc0[r];
        hbuf[(size_t)(16 + r + hi * 8) * HDIM + ncol] = (bf16_t)acc1[r];
      }
    }
    __syncthreads();  // h_t published (and x_{t+1} resident)
  }
}

// ---------------------------------------------------------------------------
// Phase B: C = A @ A for [512x512] row-major f32 (4x -> Whh^16).
// ---------------------------------------------------------------------------
__global__ __launch_bounds__(256) void
matsq_kernel(const float* __restrict__ A, float* __restrict__ C) {
  const int tid = threadIdx.x, wave = tid >> 5, lane = tid & 31;
  const int lo = lane & 15, hi = lane >> 4;
  const int tile = blockIdx.x * 8 + wave;
  const int mt = tile >> 5, nt = tile & 31;
  const int n = nt * 16 + lo;

  v8f acc;
#pragma unroll
  for (int r = 0; r < 8; ++r) acc[r] = 0.0f;

  const float* arow = A + (size_t)(mt * 16 + lo) * HDIM;
#pragma unroll
  for (int kc = 0; kc < HDIM / 32; ++kc) {
    v16bf a = frag_a_f32(arow + kc * 32 + hi * 8);
    v16bf b;  // B[k][n] = A[k][n]: column of A, stride H along k
    const float* bp = A + (size_t)(kc * 32 + hi * 8) * HDIM + n;
#pragma unroll
    for (int j = 0; j < 16; ++j) b[j] = (bf16_t)bp[(size_t)(j < 8 ? j : j + 8) * HDIM];
    acc = wmma_bf16(a, b, acc);
  }
#pragma unroll
  for (int r = 0; r < 8; ++r)
    C[(size_t)(mt * 16 + r + hi * 8) * HDIM + n] = acc[r];
}

// ---------------------------------------------------------------------------
// Phase C: sequential combine  h <- v_c + h @ (Whh^T)^16,  c = 1..NCHUNK-1.
// h carried as bf16 in LDS; final step writes f32 out from accumulators.
// ---------------------------------------------------------------------------
__global__ __launch_bounds__(1024, 1) void
combine_kernel(const unsigned* __restrict__ w16_img,  // packed Whh^16 fragments
               const float* __restrict__ v,           // [NCHUNK, B, H]
               float* __restrict__ out) {             // [B, H]
  __shared__ bf16_t hbuf[BATCH * HDIM];
  const int tid = threadIdx.x, wave = tid >> 5, lane = tid & 31;
  const int lo = lane & 15, hi = lane >> 4;
  const int ncol = wave * 16 + lo;
  const unsigned* wbase = w16_img + ((size_t)(wave * (HDIM / 32)) * 32 + lane) * 8;

  for (int i = tid; i < BATCH * HDIM; i += 1024) hbuf[i] = (bf16_t)v[i];  // h=v_0
  __syncthreads();

  for (int c = 1; c < NCHUNK; ++c) {
    const unsigned* wb = wbase + opaque_zero();  // keep W16 stream in-loop
    const float* vc = v + (size_t)c * (BATCH * HDIM);
    v8f acc0, acc1;
#pragma unroll
    for (int r = 0; r < 8; ++r) {
      acc0[r] = vc[(size_t)(r + hi * 8) * HDIM + ncol];
      acc1[r] = vc[(size_t)(16 + r + hi * 8) * HDIM + ncol];
    }
#pragma unroll
    for (int kc = 0; kc < HDIM / 32; ++kc) {
      v16bf bw = frag_b_img(wb + (size_t)kc * 256);
      v16bf a0 = frag_a_bf(hbuf + (size_t)lo * HDIM + kc * 32 + hi * 8);
      v16bf a1 = frag_a_bf(hbuf + (size_t)(lo + 16) * HDIM + kc * 32 + hi * 8);
      acc0 = wmma_bf16(a0, bw, acc0);
      acc1 = wmma_bf16(a1, bw, acc1);
    }
    __syncthreads();
    if (c == NCHUNK - 1) {
#pragma unroll
      for (int r = 0; r < 8; ++r) {
        out[(size_t)(r + hi * 8) * HDIM + ncol]      = acc0[r];
        out[(size_t)(16 + r + hi * 8) * HDIM + ncol] = acc1[r];
      }
    } else {
#pragma unroll
      for (int r = 0; r < 8; ++r) {
        hbuf[(size_t)(r + hi * 8) * HDIM + ncol]      = (bf16_t)acc0[r];
        hbuf[(size_t)(16 + r + hi * 8) * HDIM + ncol] = (bf16_t)acc1[r];
      }
    }
    __syncthreads();
  }
}

// ---------------------------------------------------------------------------
// Last-resort naive f32 scan (zero workspace); never expected to run.
// ---------------------------------------------------------------------------
__global__ __launch_bounds__(512, 1) void
naive_scan_kernel(const float* __restrict__ x, const float* __restrict__ Wxh,
                  const float* __restrict__ bxh, const float* __restrict__ Whh,
                  float* __restrict__ out) {
  __shared__ float h[BATCH * HDIM];
  __shared__ float xs[BATCH * INDIM];
  const int n = threadIdx.x;  // one output column per thread
  for (int i = n; i < BATCH * HDIM; i += 512) h[i] = 0.0f;
  for (int t = 0; t < TLEN; ++t) {
    for (int i = n; i < BATCH * INDIM; i += 512)
      xs[i] = x[(size_t)(i / INDIM) * TLEN * INDIM + (size_t)t * INDIM + (i % INDIM)];
    __syncthreads();
    float hn[BATCH];
#pragma unroll 1
    for (int b = 0; b < BATCH; ++b) {
      float acc = bxh[n];
      for (int i = 0; i < INDIM; ++i) acc += xs[b * INDIM + i] * Wxh[(size_t)n * INDIM + i];
      for (int k = 0; k < HDIM; ++k)  acc += h[b * HDIM + k] * Whh[(size_t)n * HDIM + k];
      hn[b] = acc;
    }
    __syncthreads();
    for (int b = 0; b < BATCH; ++b) h[b * HDIM + n] = hn[b];
    __syncthreads();
  }
  for (int i = n; i < BATCH * HDIM; i += 512) out[i] = h[i];
}

// ---------------------------------------------------------------------------
// Launcher
// ---------------------------------------------------------------------------
extern "C" void kernel_launch(void* const* d_in, const int* in_sizes, int n_in,
                              void* d_out, int out_size, void* d_ws, size_t ws_size,
                              hipStream_t stream) {
  (void)in_sizes; (void)n_in; (void)out_size;
  const float* x   = (const float*)d_in[0];
  const float* Wxh = (const float*)d_in[1];
  const float* bxh = (const float*)d_in[2];
  const float* Whh = (const float*)d_in[3];
  float* out = (float*)d_out;

  unsigned* ws = (unsigned*)d_ws;
  unsigned* whh_img = ws;
  unsigned* wxh_img = ws + WS_WXH_IMG;
  unsigned* xbf     = ws + WS_XBF;
  unsigned* w16_img = ws + WS_W16_IMG;
  float* p0 = (float*)(ws + WS_P0);
  float* p1 = (float*)(ws + WS_P1);
  float* v  = (float*)(ws + WS_V);
  const int x_ndw = BATCH * TLEN * INDIM / 2;  // 4194304 packed bf16 pairs

  if (d_ws != nullptr && ws_size >= (size_t)WS_FULL_DW * 4) {
    // Full chunked parallel scan.
    pack_b_image<<<64, 256, 0, stream>>>(Whh, whh_img, HDIM);
    pack_b_image<<<32, 256, 0, stream>>>(Wxh, wxh_img, INDIM);
    pack_x_bf16<<<x_ndw / 256, 256, 0, stream>>>(x, xbf, x_ndw);
    matsq_kernel<<<128, 256, 0, stream>>>(Whh, p0);  // W^2
    matsq_kernel<<<128, 256, 0, stream>>>(p0, p1);   // W^4
    matsq_kernel<<<128, 256, 0, stream>>>(p1, p0);   // W^8
    matsq_kernel<<<128, 256, 0, stream>>>(p0, p1);   // W^16
    pack_b_image<<<64, 256, 0, stream>>>(p1, w16_img, HDIM);
    chunk_scan_kernel<<<NCHUNK, 1024, 0, stream>>>(xbf, wxh_img, bxh, whh_img, v, LCHUNK);
    combine_kernel<<<1, 1024, 0, stream>>>(w16_img, v, out);
  } else if (d_ws != nullptr && ws_size >= (size_t)WS_MID_DW * 4) {
    // Single-workgroup WMMA scan over the full sequence.
    pack_b_image<<<64, 256, 0, stream>>>(Whh, whh_img, HDIM);
    pack_b_image<<<32, 256, 0, stream>>>(Wxh, wxh_img, INDIM);
    pack_x_bf16<<<x_ndw / 256, 256, 0, stream>>>(x, xbf, x_ndw);
    chunk_scan_kernel<<<1, 1024, 0, stream>>>(xbf, wxh_img, bxh, whh_img, out, TLEN);
  } else {
    naive_scan_kernel<<<1, 512, 0, stream>>>(x, Wxh, bxh, Whh, out);
  }
}
